// GMSRModel_55937654063572
// MI455X (gfx1250) — compile-verified, matrix operand order checked by reading, not run on hardware
//
#include <hip/hip_runtime.h>
#include <hip/hip_bf16.h>
#include <math.h>

// ---------------------------------------------------------------------------
// GMSR model for MI455X (gfx1250, wave32, WMMA).
//
// d_out layout: h_return (1024*256*128 f32) then H (4*256*128 f32).
// d_ws: first 512 KB = W repacked to f16 in WMMA B-fragment-major layout.
//
//  * emb kernel  : (relu(X@FC_W+b))@FC_W2+b2 -> written into h_return region;
//                  A and B tiles staged in LDS in fragment-major f16 so every
//                  WMMA operand is a single 32B LDS vector load.
//  * scan kernel : 16 WGs x 16 batch rows; each WG runs all 1024 timesteps
//                  privately (batch is the only parallel axis of the
//                  recurrence -> zero inter-WG sync). State kept in LDS with a
//                  rotating 128-column block base (no physical shift), plus an
//                  A-fragment-major f16 mirror refreshed in the cs-multiply
//                  pass. B operands stream from the L2-resident packed W.
// ---------------------------------------------------------------------------

typedef __attribute__((ext_vector_type(16))) _Float16 v16h;
typedef __attribute__((ext_vector_type(8)))  float    v8f;

#define SEQ   1024
#define BATCH 256
#define HDIM  128
#define KDIM  4
#define KH    512                       // KDIM * HDIM
#define HRET_ELEMS (SEQ * BATCH * HDIM) // 33554432

// forward map: in-slice k for element e of a lane's v16h (ISA 7.12.2, 16-bit)
__device__ __forceinline__ int kmap(int e, int lane) {
    return e + ((e >= 8) ? 8 : 0) + ((lane >= 16) ? 8 : 0);
}
// inverse map: in-slice k (0..31) -> (element e, lane-half hi)
__device__ __forceinline__ void kinv(int kk, int& e, int& hi) {
    hi = (kk >> 3) & 1;
    e  = (kk & 7) | ((kk & 16) >> 1);
}

// ---------------------------------------------------------------------------
// Kernel 0: repack W (512x512 f32 row-major) into f16 B-fragment-major:
//   W16f[((ks*32 + nt)*32 + lane)*16 + e] = (f16) W[(ks*32+kmap(e,lane))*512
//                                                   + nt*16 + (lane&15)]
// ---------------------------------------------------------------------------
__global__ __launch_bounds__(256) void prep_w_kernel(const float* __restrict__ W,
                                                     _Float16* __restrict__ W16f) {
    int idx  = blockIdx.x * 256 + threadIdx.x;  // 0 .. 512*512-1
    int e    = idx & 15;
    int lane = (idx >> 4) & 31;
    int nt   = (idx >> 9) & 31;
    int ks   = idx >> 14;
    int k = ks * 32 + kmap(e, lane);
    int n = nt * 16 + (lane & 15);
    W16f[idx] = (_Float16)W[k * KH + n];
}

// ---------------------------------------------------------------------------
// Kernel 1: inputs_emb = relu(X@FC_W + FC_B) @ FC_W2 + FC_B2  (64 rows/block)
// ---------------------------------------------------------------------------
__global__ __launch_bounds__(256) void emb_kernel(const float* __restrict__ X,
                                                  const float* __restrict__ FC_W,
                                                  const float* __restrict__ FC_B,
                                                  const float* __restrict__ FC_W2,
                                                  const float* __restrict__ FC_B2,
                                                  float* __restrict__ out) {
    __shared__ _Float16 Af[64 * 128];    // A fragments (frag-major)  16 KB
    __shared__ _Float16 Wf[128 * 128];   // B fragments (frag-major)  32 KB

    const int tid  = threadIdx.x;
    const int lane = tid & 31;
    const int w    = tid >> 5;            // 8 waves
    const int r0   = blockIdx.x * 64;

    // stage A fragment-major: frag id = mt*4+ks (mt:4, ks:4)
    for (int i = tid; i < 64 * 128; i += 256) {
        int e = i & 15, ln = (i >> 4) & 31, fid = i >> 9;
        int ks = fid & 3, mt = fid >> 2;
        int m = mt * 16 + (ln & 15);
        int k = ks * 32 + kmap(e, ln);
        Af[i] = (_Float16)X[(size_t)(r0 + m) * 128 + k];
    }
    // stage FC_W fragment-major: frag id = ks*8+nt (ks:4, nt:8)
    for (int i = tid; i < 128 * 128; i += 256) {
        int e = i & 15, ln = (i >> 4) & 31, fid = i >> 9;
        int nt = fid & 7, ks = fid >> 3;
        int k = ks * 32 + kmap(e, ln);
        int n = nt * 16 + (ln & 15);
        Wf[i] = (_Float16)FC_W[k * 128 + n];
    }
    __syncthreads();

    const int mt  = w & 3;               // m-tile 0..3
    const int nh  = w >> 2;              // n-half 0..1 (n-tiles nh*4..nh*4+3)
    const int khi = (lane >= 16) ? 8 : 0;
    const int l15 = lane & 15;

    // ---- GEMM 1 ----
    v8f acc[4];
    #pragma unroll
    for (int nt = 0; nt < 4; ++nt) acc[nt] = {};
    #pragma unroll
    for (int ks = 0; ks < 4; ++ks) {
        const v16h a = *(const v16h*)(Af + (((mt * 4 + ks) * 32 + lane) << 4));
        #pragma unroll
        for (int nt = 0; nt < 4; ++nt) {
            const v16h b = *(const v16h*)(Wf + (((ks * 8 + nh * 4 + nt) * 32 + lane) << 4));
            acc[nt] = __builtin_amdgcn_wmma_f32_16x16x32_f16(
                false, a, false, b, (short)0, acc[nt], false, false);
        }
    }
    __syncthreads();

    // relu(+bias) -> back into Af in fragment-major; stage FC_W2 over Wf
    #pragma unroll
    for (int nt = 0; nt < 4; ++nt) {
        const int col = (nh * 4 + nt) * 16 + l15;
        const float bb = FC_B[col];
        const int ks2 = col >> 5;
        int e2, hi2; kinv(col & 31, e2, hi2);
        #pragma unroll
        for (int e = 0; e < 8; ++e) {
            float v = fmaxf(acc[nt][e] + bb, 0.0f);
            int m_in = e + khi;
            Af[(((mt * 4 + ks2) * 32 + hi2 * 16 + m_in) << 4) + e2] = (_Float16)v;
        }
    }
    for (int i = tid; i < 128 * 128; i += 256) {
        int e = i & 15, ln = (i >> 4) & 31, fid = i >> 9;
        int nt = fid & 7, ks = fid >> 3;
        int k = ks * 32 + kmap(e, ln);
        int n = nt * 16 + (ln & 15);
        Wf[i] = (_Float16)FC_W2[k * 128 + n];
    }
    __syncthreads();

    // ---- GEMM 2 ----
    v8f acc2[4];
    #pragma unroll
    for (int nt = 0; nt < 4; ++nt) acc2[nt] = {};
    #pragma unroll
    for (int ks = 0; ks < 4; ++ks) {
        const v16h a = *(const v16h*)(Af + (((mt * 4 + ks) * 32 + lane) << 4));
        #pragma unroll
        for (int nt = 0; nt < 4; ++nt) {
            const v16h b = *(const v16h*)(Wf + (((ks * 8 + nh * 4 + nt) * 32 + lane) << 4));
            acc2[nt] = __builtin_amdgcn_wmma_f32_16x16x32_f16(
                false, a, false, b, (short)0, acc2[nt], false, false);
        }
    }
    #pragma unroll
    for (int nt = 0; nt < 4; ++nt) {
        const int col = (nh * 4 + nt) * 16 + l15;
        const float bb = FC_B2[col];
        #pragma unroll
        for (int e = 0; e < 8; ++e)
            out[(size_t)(r0 + mt * 16 + e + khi) * 128 + col] = acc2[nt][e] + bb;
    }
}

// ---------------------------------------------------------------------------
// Kernel 2: the scan. gridDim.x = 16 WGs, blockDim.x = 512 (16 waves).
// Physical state blocks rotate: at step t (bb = t&3), logical block j lives at
// physical block (bb+j)&3; h_out is written into physical block bb.
// ---------------------------------------------------------------------------
__global__ __launch_bounds__(512, 1) void scan_kernel(const float* __restrict__ hidden,
                                                      const float* __restrict__ Rv,
                                                      const float* __restrict__ Bv,
                                                      const _Float16* __restrict__ W16f,
                                                      float* __restrict__ out) {
    __shared__ float    preH[16 * KH];     // f32 master state           32 KB
    __shared__ _Float16 preH16[16 * KH];   // A-fragment-major mirror    16 KB
    __shared__ _Float16 attS[16 * KH];     // att (f16)                  16 KB

    const int tid  = threadIdx.x;
    const int lane = tid & 31;
    const int w    = tid >> 5;             // 16 waves; wave w -> n-tiles 2w,2w+1
    const int rb0  = blockIdx.x * 16;
    const int khi  = (lane >= 16) ? 8 : 0;
    const int l15  = lane & 15;

    // per-thread physical column (blockDim == KH), its fragment coordinates
    const int pc  = tid;                   // physical column owned by thread
    const int pb  = pc >> 7;               // physical block 0..3
    const int c0  = pc & 127;              // column within block
    const int ksp = pc >> 5;               // physical 32-wide k-slice
    int efr, hfr; kinv(pc & 31, efr, hfr);
    const int fbase = ksp * 512 + hfr * 256 + efr;   // + r*16 per row

    // cs factor for this column under each bb rotation: d[u] = cs[((pb-u)&3), c0]
    float d[4];
    #pragma unroll
    for (int u = 0; u < 4; ++u) {
        int lb = (pb - u) & 3;
        float r = Rv[lb * 64 + (c0 & 63)];
        d[u] = (c0 < 64) ? __cosf(r) : __sinf(r);
    }
    // bias for this wave's two output column groups
    float biasreg[2];
    #pragma unroll
    for (int j = 0; j < 2; ++j) biasreg[j] = Bv[(2 * w + j) * 16 + l15];

    // preH0 = tanh(transpose(hidden,(1,0,2)).reshape(batch,kh)); bb=0 at t=0
    for (int i = tid; i < 16 * KH; i += 512) {
        int r = i >> 9, col = i & (KH - 1);
        int kk = col >> 7, c = col & 127;
        preH[i] = tanhf(hidden[(size_t)(kk * BATCH + rb0 + r) * HDIM + c]);
    }
    __syncthreads();

    for (int t4 = 0; t4 < SEQ; t4 += 4) {
        #pragma unroll
        for (int u = 0; u < 4; ++u) {      // bb == u (compile-time constant)
            const int t = t4 + u;

            // ---- 1. p3 = preH * cs (own column), refresh f16 frag mirror ----
            {
                const float cs = d[u];
                #pragma unroll
                for (int r = 0; r < 16; ++r) {
                    float v = preH[r * KH + pc] * cs;
                    preH[r * KH + pc] = v;
                    preH16[fbase + r * 16] = (_Float16)v;
                }
            }
            __syncthreads();

            // ---- 2. att = p3 @ W + B (16 x 512, K=512) ----
            v8f acc0 = {}, acc1 = {};
            #pragma unroll
            for (int ks = 0; ks < 16; ++ks) {          // logical k-slice
                const int psl = (((u + (ks >> 2)) & 3) << 2) | (ks & 3);
                const v16h a  = *(const v16h*)(preH16 + ((psl * 32 + lane) << 4));
                const v16h b0 = *(const v16h*)(W16f + ((size_t)((ks * 32 + 2 * w + 0) * 32 + lane) << 4));
                const v16h b1 = *(const v16h*)(W16f + ((size_t)((ks * 32 + 2 * w + 1) * 32 + lane) << 4));
                acc0 = __builtin_amdgcn_wmma_f32_16x16x32_f16(false, a, false, b0,
                                                              (short)0, acc0, false, false);
                acc1 = __builtin_amdgcn_wmma_f32_16x16x32_f16(false, a, false, b1,
                                                              (short)0, acc1, false, false);
            }
            #pragma unroll
            for (int e = 0; e < 8; ++e) {
                int r = e + khi;
                attS[r * KH + (2 * w + 0) * 16 + l15] = (_Float16)(acc0[e] + biasreg[0]);
                attS[r * KH + (2 * w + 1) * 16 + l15] = (_Float16)(acc1[e] + biasreg[1]);
            }
            __syncthreads();

            // ---- 3. softmax over k of |att/2|, weighted sum, tanh ----
            float hq[4];
            #pragma unroll
            for (int q = 0; q < 4; ++q) {              // 16*128 / 512 threads
                int i = tid + q * 512;
                int r = i >> 7, c = i & 127;
                float a0 = fabsf((float)attS[r * KH + 0 * 128 + c]) * 0.5f;
                float a1 = fabsf((float)attS[r * KH + 1 * 128 + c]) * 0.5f;
                float a2 = fabsf((float)attS[r * KH + 2 * 128 + c]) * 0.5f;
                float a3 = fabsf((float)attS[r * KH + 3 * 128 + c]) * 0.5f;
                float mx = fmaxf(fmaxf(a0, a1), fmaxf(a2, a3));
                float e0 = __expf(a0 - mx), e1 = __expf(a1 - mx);
                float e2 = __expf(a2 - mx), e3 = __expf(a3 - mx);
                float den = e0 + e1 + e2 + e3;
                // p3 logical block j at physical block (u+j)&3
                float pin = (preH[r * KH + (((u + 0) & 3) << 7) + c] * e0 +
                             preH[r * KH + (((u + 1) & 3) << 7) + c] * e1 +
                             preH[r * KH + (((u + 2) & 3) << 7) + c] * e2 +
                             preH[r * KH + (((u + 3) & 3) << 7) + c] * e3) / den;
                size_t oidx = (size_t)(t * BATCH + rb0 + r) * HDIM + c;
                float x = out[oidx];                   // inputs_emb (from emb_kernel)
                float h = tanhf(x * 0.125f + pin);
                out[oidx] = h;                         // h_return[t]
                if (t >= SEQ - KDIM)                   // H = h_return[-k:]
                    out[(size_t)HRET_ELEMS +
                        (size_t)((t - (SEQ - KDIM)) * BATCH + rb0 + r) * HDIM + c] = h;
                hq[q] = h;
            }
            __syncthreads();                           // all p3 reads done

            // ---- 4. logical shift == write h into physical block bb(=u) ----
            #pragma unroll
            for (int q = 0; q < 4; ++q) {
                int i = tid + q * 512;
                int r = i >> 7, c = i & 127;
                preH[r * KH + (u << 7) + c] = hq[q];
            }
            __syncthreads();
        }
    }
}

// ---------------------------------------------------------------------------
extern "C" void kernel_launch(void* const* d_in, const int* in_sizes, int n_in,
                              void* d_out, int out_size, void* d_ws, size_t ws_size,
                              hipStream_t stream) {
    (void)in_sizes; (void)n_in; (void)out_size; (void)ws_size;
    const float* inputs = (const float*)d_in[0];   // (1024,256,128)
    const float* hidden = (const float*)d_in[1];   // (4,256,128)
    const float* W      = (const float*)d_in[2];   // (512,512)
    const float* B      = (const float*)d_in[3];   // (1,512)
    const float* R      = (const float*)d_in[4];   // (1,256)
    const float* FC_W   = (const float*)d_in[5];   // (128,128)
    const float* FC_B   = (const float*)d_in[6];   // (1,128)
    const float* FC_W2  = (const float*)d_in[7];   // (128,128)
    const float* FC_B2  = (const float*)d_in[8];   // (1,128)
    float* out = (float*)d_out;
    _Float16* W16f = (_Float16*)d_ws;              // 512 KB packed W

    prep_w_kernel<<<(KH * KH) / 256, 256, 0, stream>>>(W, W16f);
    emb_kernel<<<(SEQ * BATCH) / 64, 256, 0, stream>>>(inputs, FC_W, FC_B,
                                                       FC_W2, FC_B2, out);
    scan_kernel<<<BATCH / 16, 512, 0, stream>>>(hidden, R, B, W16f, out);
}
